// PropagationGCN_50611894616877
// MI455X (gfx1250) — matrix operand order repaired and checked by reading, not compiled, not run on hardware
//
#include <hip/hip_runtime.h>

// ---------------- types ----------------
typedef __attribute__((ext_vector_type(8)))  float  v8f;
typedef __attribute__((ext_vector_type(16))) __bf16 v16bf;
typedef __attribute__((ext_vector_type(8)))  __bf16 v8bf;
typedef unsigned int u32x4 __attribute__((ext_vector_type(4)));
typedef int          i32x8 __attribute__((ext_vector_type(8)));
typedef int          i32x4 __attribute__((ext_vector_type(4)));

#define HD 256   // hidden channels
#define IND 768  // in channels

// =====================================================================
// TDM helper: 2D tile load (32 k x 128 lines, bf16) into LDS with
// 16B padding every 64B row (LDS row stride = 40 bf16).
// Descriptor per CDNA5 ISA 8.3/8.4: count=1, type=2, data_size=2B.
// =====================================================================
__device__ __forceinline__
void tdm_tile_load(unsigned lds_off, const void* gptr,
                   unsigned tensor_d0, unsigned tensor_d1,
                   unsigned long long stride0) {
    unsigned long long ga = (unsigned long long)gptr;
    u32x4 g0;
    g0[0] = 1u;                                    // count=1 (user descriptor)
    g0[1] = lds_off;                               // lds_addr (bytes)
    g0[2] = (unsigned)ga;                          // global_addr[31:0]
    g0[3] = (unsigned)((ga >> 32) & 0x01ffffffu)   // global_addr[56:32]
          | (2u << 30);                            // type = 2 ("image")
    i32x8 g1;
    g1[0] = (int)((1u << 16)    // data_size = 2 bytes
                | (1u << 20)    // pad_enable
                | (3u << 22)    // pad_interval: 16 DWORDs (64B tile row)
                | (3u << 25));  // pad_amount:   4 DWORDs (16B)
    g1[1] = (int)((tensor_d0 & 0xffffu) << 16);                  // dim0[15:0]
    g1[2] = (int)((tensor_d0 >> 16) | ((tensor_d1 & 0xffffu) << 16));
    g1[3] = (int)((tensor_d1 >> 16) | (32u << 16));              // tile_dim0 = 32
    g1[4] = 128;                                                 // tile_dim1 = 128
    g1[5] = (int)(stride0 & 0xffffffffull);                      // dim0 stride lo
    g1[6] = (int)((stride0 >> 32) & 0xffffull);                  // stride hi
    g1[7] = 0;
    const i32x4 z4 = {0, 0, 0, 0};
#if __clang_major__ >= 23
    const i32x8 z8 = {0, 0, 0, 0, 0, 0, 0, 0};
    __builtin_amdgcn_tensor_load_to_lds(g0, g1, z4, z4, z8, 0);
#else
    __builtin_amdgcn_tensor_load_to_lds(g0, g1, z4, z4, 0);
#endif
}

// =====================================================================
// GEMM1: C[Nrows,256] = A_f32[Nrows,K] * B_f32[K,256]; fused fp32->bf16
// staging (x is read once from HBM), bf16 WMMA, fp32 accumulate.
// =====================================================================
__global__ __launch_bounds__(256)
void gemm_bf16_wmma(const float* __restrict__ A, const float* __restrict__ B,
                    float* __restrict__ C, int Nrows, int K) {
    const int tid  = threadIdx.x;
    const int lane = tid & 31;
    const int wid  = tid >> 5;
    const int waveM = wid & 3;
    const int waveN = wid >> 2;
    const int mlan  = lane & 15;
    const int half  = lane >> 4;

    const int rowBase = blockIdx.y * 128;
    const int colBase = blockIdx.x * 128;

    __shared__ __align__(16) __bf16 As[128][40];
    __shared__ __align__(16) __bf16 Bs[128][40];

    v8f acc[2][4];
    const v8f vzero = {0.f,0.f,0.f,0.f,0.f,0.f,0.f,0.f};
#pragma unroll
    for (int i = 0; i < 2; ++i)
#pragma unroll
        for (int j = 0; j < 4; ++j) acc[i][j] = vzero;

    for (int k0 = 0; k0 < K; k0 += 32) {
        {   // stage A tile 128x32 (fp32 -> bf16)
            const int tr = tid >> 3;
            const int tc = (tid & 7) << 2;
#pragma unroll
            for (int p = 0; p < 4; ++p) {
                const int r = p * 32 + tr;
                const int grow = rowBase + r;
                float4 v = {0.f, 0.f, 0.f, 0.f};
                if (grow < Nrows)
                    v = *(const float4*)(A + (size_t)grow * K + k0 + tc);
                As[r][tc + 0] = (__bf16)v.x;
                As[r][tc + 1] = (__bf16)v.y;
                As[r][tc + 2] = (__bf16)v.z;
                As[r][tc + 3] = (__bf16)v.w;
            }
        }
        {   // stage B tile 32x128 transposed -> [n][k]
            const int bk = tid >> 5;
            const int bc = (tid & 31) << 2;
#pragma unroll
            for (int p = 0; p < 4; ++p) {
                const int k = p * 8 + bk;
                float4 v = *(const float4*)(B + (size_t)(k0 + k) * HD + colBase + bc);
                Bs[bc + 0][k] = (__bf16)v.x;
                Bs[bc + 1][k] = (__bf16)v.y;
                Bs[bc + 2][k] = (__bf16)v.z;
                Bs[bc + 3][k] = (__bf16)v.w;
            }
        }
        if (k0 + 32 < K && tid < 128) {          // global_prefetch_b8
            const int grow = rowBase + tid;
            if (grow < Nrows)
                __builtin_prefetch(A + (size_t)grow * K + k0 + 32, 0, 1);
        }
        __syncthreads();

        v16bf afr[2];
#pragma unroll
        for (int fm = 0; fm < 2; ++fm) {
            const int r = waveM * 32 + fm * 16 + mlan;
            v8bf lo = *(const v8bf*)&As[r][half * 8];
            v8bf hi = *(const v8bf*)&As[r][16 + half * 8];
            afr[fm] = __builtin_shufflevector(lo, hi,
                0,1,2,3,4,5,6,7,8,9,10,11,12,13,14,15);
        }
        v16bf bfr[4];
#pragma unroll
        for (int fn = 0; fn < 4; ++fn) {
            const int cn = waveN * 64 + fn * 16 + mlan;
            v8bf lo = *(const v8bf*)&Bs[cn][half * 8];
            v8bf hi = *(const v8bf*)&Bs[cn][16 + half * 8];
            bfr[fn] = __builtin_shufflevector(lo, hi,
                0,1,2,3,4,5,6,7,8,9,10,11,12,13,14,15);
        }
#pragma unroll
        for (int fm = 0; fm < 2; ++fm)
#pragma unroll
            for (int fn = 0; fn < 4; ++fn)
                acc[fm][fn] = __builtin_amdgcn_wmma_f32_16x16x32_bf16(
                    false, afr[fm], false, bfr[fn],
                    (short)0, acc[fm][fn], false, false);
        __syncthreads();
    }

#pragma unroll
    for (int fm = 0; fm < 2; ++fm)
#pragma unroll
        for (int fn = 0; fn < 4; ++fn) {
            const int cn = colBase + waveN * 64 + fn * 16 + mlan;
#pragma unroll
            for (int v = 0; v < 8; ++v) {
                const int grow = rowBase + waveM * 32 + fm * 16 + half * 8 + v;
                if (grow < Nrows)
                    C[(size_t)grow * HD + cn] = acc[fm][fn][v];
            }
        }
}

// =====================================================================
// GEMM2 / root GEMM: C[Nrows,256] = A_bf16[Nrows,K] * Bt_bf16[256,K]^T
// TDM double-buffered: wave 0 DMAs next (A,B) tile pair while all waves
// run WMMA on the current pair. TENSORcnt is in-order per wave, so
// s_wait_tensorcnt(2) after issuing the next pair => current pair done.
// =====================================================================
__global__ __launch_bounds__(256)
void gemm_tdm_wmma(const __bf16* __restrict__ A, const __bf16* __restrict__ Bt,
                   float* __restrict__ C, int Nrows, int K) {
    const int tid  = threadIdx.x;
    const int lane = tid & 31;
    const int wid  = tid >> 5;
    const int waveM = wid & 3;
    const int waveN = wid >> 2;
    const int mlan  = lane & 15;
    const int half  = lane >> 4;

    const int rowBase = blockIdx.y * 128;
    const int colBase = blockIdx.x * 128;

    // TDM writes 64B rows + 16B pad -> row stride 40 bf16, 16B aligned.
    __shared__ __align__(16) __bf16 As[2][128][40];
    __shared__ __align__(16) __bf16 Bs[2][128][40];

    v8f acc[2][4];
    const v8f vzero = {0.f,0.f,0.f,0.f,0.f,0.f,0.f,0.f};
#pragma unroll
    for (int i = 0; i < 2; ++i)
#pragma unroll
        for (int j = 0; j < 4; ++j) acc[i][j] = vzero;

    const int nk = K >> 5;

    if (wid == 0) {   // prime buffer 0
        tdm_tile_load((unsigned)(size_t)&As[0][0][0],
                      A + (size_t)rowBase * K, (unsigned)K, (unsigned)Nrows,
                      (unsigned long long)K);
        tdm_tile_load((unsigned)(size_t)&Bs[0][0][0],
                      Bt + (size_t)colBase * K, (unsigned)K, (unsigned)HD,
                      (unsigned long long)K);
    }

    for (int kt = 0; kt < nk; ++kt) {
        const int cur = kt & 1;
        if (wid == 0) {
            if (kt + 1 < nk) {
                const int k0n = (kt + 1) << 5;
                tdm_tile_load((unsigned)(size_t)&As[cur ^ 1][0][0],
                              A + (size_t)rowBase * K + k0n, (unsigned)K,
                              (unsigned)Nrows, (unsigned long long)K);
                tdm_tile_load((unsigned)(size_t)&Bs[cur ^ 1][0][0],
                              Bt + (size_t)colBase * K + k0n, (unsigned)K,
                              (unsigned)HD, (unsigned long long)K);
                __builtin_amdgcn_s_wait_tensorcnt(2);  // current pair landed
            } else {
                __builtin_amdgcn_s_wait_tensorcnt(0);
            }
        }
        __syncthreads();

        v16bf afr[2];
#pragma unroll
        for (int fm = 0; fm < 2; ++fm) {
            const int r = waveM * 32 + fm * 16 + mlan;
            v8bf lo = *(const v8bf*)&As[cur][r][half * 8];
            v8bf hi = *(const v8bf*)&As[cur][r][16 + half * 8];
            afr[fm] = __builtin_shufflevector(lo, hi,
                0,1,2,3,4,5,6,7,8,9,10,11,12,13,14,15);
        }
        v16bf bfr[4];
#pragma unroll
        for (int fn = 0; fn < 4; ++fn) {
            const int cn = waveN * 64 + fn * 16 + mlan;
            v8bf lo = *(const v8bf*)&Bs[cur][cn][half * 8];
            v8bf hi = *(const v8bf*)&Bs[cur][cn][16 + half * 8];
            bfr[fn] = __builtin_shufflevector(lo, hi,
                0,1,2,3,4,5,6,7,8,9,10,11,12,13,14,15);
        }
#pragma unroll
        for (int fm = 0; fm < 2; ++fm)
#pragma unroll
            for (int fn = 0; fn < 4; ++fn)
                acc[fm][fn] = __builtin_amdgcn_wmma_f32_16x16x32_bf16(
                    false, afr[fm], false, bfr[fn],
                    (short)0, acc[fm][fn], false, false);
        __syncthreads();
    }

#pragma unroll
    for (int fm = 0; fm < 2; ++fm)
#pragma unroll
        for (int fn = 0; fn < 4; ++fn) {
            const int cn = colBase + waveN * 64 + fn * 16 + mlan;
#pragma unroll
            for (int v = 0; v < 8; ++v) {
                const int grow = rowBase + waveM * 32 + fm * 16 + half * 8 + v;
                if (grow < Nrows)
                    C[(size_t)grow * HD + cn] = acc[fm][fn][v];
            }
        }
}

// =====================================================================
// Graph-structure kernels
// =====================================================================
__global__ void deg_init_kernel(float* deg, int N) {
    int n = blockIdx.x * blockDim.x + threadIdx.x;
    if (n < N) deg[n] = 1.0f;   // self loop
}

__global__ void deg_edge_kernel(const long long* __restrict__ dst, float* deg, int E) {
    int e = blockIdx.x * blockDim.x + threadIdx.x;
    if (e < E) atomicAdd(&deg[(int)dst[e]], 1.0f);
}

__global__ void dinv_kernel(float* deg, int N) {
    int n = blockIdx.x * blockDim.x + threadIdx.x;
    if (n < N) deg[n] = rsqrtf(deg[n]);
}

__global__ void roots_kernel(const long long* __restrict__ batch, int* roots, int N) {
    int n = blockIdx.x * blockDim.x + threadIdx.x;
    if (n >= N) return;
    if (n == 0 || batch[n] != batch[n - 1]) roots[(int)batch[n]] = n;
}

// Wt[n][k] = (bf16)W[k][n]
__global__ void wt_bf16_kernel(const float* __restrict__ W, __bf16* __restrict__ Wt, int K) {
    int t = blockIdx.x * blockDim.x + threadIdx.x;
    if (t >= K * HD) return;
    const int k = t >> 8;
    const int n = t & 255;
    Wt[(size_t)n * K + k] = (__bf16)W[t];
}

// agg[n][c] = dinv[n]^2 * xw[n][c] + b[c]
__global__ void agg_init_kernel(const float* __restrict__ xw, const float* __restrict__ dinv,
                                const float* __restrict__ b, float* __restrict__ agg, int N) {
    int t = blockIdx.x * blockDim.x + threadIdx.x;
    if (t >= N * 64) return;
    const int n = t >> 6;
    const int c = (t & 63) << 2;
    float s = dinv[n]; s *= s;
    float4 v = *(const float4*)(xw + (size_t)n * HD + c);
    float4 o;
    o.x = s * v.x + b[c + 0];
    o.y = s * v.y + b[c + 1];
    o.z = s * v.z + b[c + 2];
    o.w = s * v.w + b[c + 3];
    *(float4*)(agg + (size_t)n * HD + c) = o;
}

// agg[dst] += dinv[src]*dinv[dst] * xw[src]
__global__ void scatter_kernel(const long long* __restrict__ src, const long long* __restrict__ dst,
                               const float* __restrict__ dinv, const float* __restrict__ xw,
                               float* __restrict__ agg, int E) {
    int t = blockIdx.x * blockDim.x + threadIdx.x;
    if (t >= E * 64) return;
    const int e = t >> 6;
    const int c = (t & 63) << 2;
    const int s = (int)src[e];
    const int d = (int)dst[e];
    const float nrm = dinv[s] * dinv[d];
    float4 v = *(const float4*)(xw + (size_t)s * HD + c);
    float* out = agg + (size_t)d * HD + c;
    atomicAdd(out + 0, nrm * v.x);
    atomicAdd(out + 1, nrm * v.y);
    atomicAdd(out + 2, nrm * v.z);
    atomicAdd(out + 3, nrm * v.w);
}

// h1bf = relu(agg) as bf16 (h1 is only consumed by GEMM2 -> halve traffic)
__global__ void relu_bf16_kernel(const float* __restrict__ agg, __bf16* __restrict__ hbf, int N) {
    int t = blockIdx.x * blockDim.x + threadIdx.x;
    if (t >= N * 64) return;
    float4 v = *((const float4*)agg + t);
    __bf16* o = hbf + (size_t)t * 4;
    o[0] = (__bf16)fmaxf(v.x, 0.f);
    o[1] = (__bf16)fmaxf(v.y, 0.f);
    o[2] = (__bf16)fmaxf(v.z, 0.f);
    o[3] = (__bf16)fmaxf(v.w, 0.f);
}

__global__ void relu_kernel(float* h, int N) {
    int t = blockIdx.x * blockDim.x + threadIdx.x;
    if (t >= N * 64) return;
    float4* p = (float4*)(h) + t;
    float4 v = *p;
    v.x = fmaxf(v.x, 0.f); v.y = fmaxf(v.y, 0.f);
    v.z = fmaxf(v.z, 0.f); v.w = fmaxf(v.w, 0.f);
    *p = v;
}

// xrbf[g][:] = (bf16)x[roots[g]][:]
__global__ void gather_roots_bf16_kernel(const float* __restrict__ x, const int* __restrict__ roots,
                                         __bf16* __restrict__ xrbf, int G) {
    int t = blockIdx.x * blockDim.x + threadIdx.x;
    if (t >= G * (IND / 4)) return;
    const int g  = t / (IND / 4);
    const int c4 = (t % (IND / 4)) * 4;
    float4 v = *(const float4*)(x + (size_t)roots[g] * IND + c4);
    __bf16* o = xrbf + (size_t)g * IND + c4;
    o[0] = (__bf16)v.x; o[1] = (__bf16)v.y;
    o[2] = (__bf16)v.z; o[3] = (__bf16)v.w;
}

__global__ __launch_bounds__(HD)
void pool_kernel(const float* __restrict__ h, const int* __restrict__ roots,
                 float* __restrict__ hpool, int N, int G) {
    const int g = blockIdx.x;
    const int c = threadIdx.x;
    const int start = roots[g];
    const int end   = (g + 1 < G) ? roots[g + 1] : N;
    float m = -3.402823466e38f;
    for (int n = start; n < end; ++n)
        m = fmaxf(m, h[(size_t)n * HD + c]);
    hpool[(size_t)g * HD + c] = m;
}

__global__ void final_kernel(const float* __restrict__ hpool, const float* __restrict__ xrw,
                             const float* __restrict__ br, const float* __restrict__ Wo,
                             const float* __restrict__ bo, float* __restrict__ out, int G) {
    int t = blockIdx.x * blockDim.x + threadIdx.x;
    if (t >= G * 2) return;
    const int g = t >> 1;
    const int c = t & 1;
    const float* hp = hpool + (size_t)g * HD;
    const float* hr = xrw   + (size_t)g * HD;
    float acc = bo[c];
    for (int k = 0; k < HD; ++k) acc += hp[k] * Wo[k * 2 + c];
    for (int k = 0; k < HD; ++k) {
        float v = hr[k] + br[k];
        acc += fmaxf(v, 0.f) * Wo[(HD + k) * 2 + c];
    }
    out[t] = acc;
}

// =====================================================================
extern "C" void kernel_launch(void* const* d_in, const int* in_sizes, int n_in,
                              void* d_out, int out_size, void* d_ws, size_t ws_size,
                              hipStream_t stream) {
    const float*     x     = (const float*)d_in[0];
    const long long* ei    = (const long long*)d_in[1];
    const long long* batch = (const long long*)d_in[2];
    const float* W1 = (const float*)d_in[3];
    const float* b1 = (const float*)d_in[4];
    const float* W2 = (const float*)d_in[5];
    const float* b2 = (const float*)d_in[6];
    const float* Wr = (const float*)d_in[7];
    const float* br = (const float*)d_in[8];
    const float* Wo = (const float*)d_in[9];
    const float* bo = (const float*)d_in[10];
    float* out = (float*)d_out;

    const int N = in_sizes[0] / IND;
    const int E = in_sizes[1] / 2;
    const int G = out_size / 2;
    const long long* esrc = ei;
    const long long* edst = ei + E;

    // -------- workspace partition --------
    float*  ws    = (float*)d_ws;
    float*  xw    = ws;                               // N*256 f32
    float*  agg   = xw + (size_t)N * HD;              // N*256 f32
    float*  dinv  = agg + (size_t)N * HD;             // N f32
    __bf16* h1bf  = (__bf16*)(dinv + N);              // N*256 bf16
    __bf16* w2t   = h1bf + (size_t)N * HD;            // 256*256 bf16
    __bf16* wrt   = w2t + (size_t)HD * HD;            // 256*768 bf16
    __bf16* xrbf  = wrt + (size_t)HD * IND;           // G*768 bf16
    float*  xrw   = (float*)(xrbf + (size_t)G * IND); // G*256 f32
    float*  hpool = xrw + (size_t)G * HD;             // G*256 f32
    int*    roots = (int*)(hpool + (size_t)G * HD);   // G i32

    const int TB = 256;
    const dim3 blk(TB);

    // weights -> bf16 transposed [n][k]
    wt_bf16_kernel<<<dim3((HD * HD + TB - 1) / TB), blk, 0, stream>>>(W2, w2t, HD);
    wt_bf16_kernel<<<dim3((IND * HD + TB - 1) / TB), blk, 0, stream>>>(Wr, wrt, IND);

    // degrees -> dinv; roots
    deg_init_kernel<<<dim3((N + TB - 1) / TB), blk, 0, stream>>>(dinv, N);
    deg_edge_kernel<<<dim3((E + TB - 1) / TB), blk, 0, stream>>>(edst, dinv, E);
    dinv_kernel<<<dim3((N + TB - 1) / TB), blk, 0, stream>>>(dinv, N);
    roots_kernel<<<dim3((N + TB - 1) / TB), blk, 0, stream>>>(batch, roots, N);

    const dim3 gemmGridN(HD / 128, (N + 127) / 128);
    const dim3 gemmGridG(HD / 128, (G + 127) / 128);
    const dim3 nc((N * 64 + TB - 1) / TB);
    const dim3 ec((E * 64 + TB - 1) / TB);

    // ---- layer 1: fused-convert WMMA GEMM ----
    gemm_bf16_wmma<<<gemmGridN, blk, 0, stream>>>(x, W1, xw, N, IND);
    agg_init_kernel<<<nc, blk, 0, stream>>>(xw, dinv, b1, agg, N);
    scatter_kernel<<<ec, blk, 0, stream>>>(esrc, edst, dinv, xw, agg, E);
    relu_bf16_kernel<<<nc, blk, 0, stream>>>(agg, h1bf, N);     // h1 (bf16)

    // ---- layer 2: TDM double-buffered WMMA GEMM ----
    gemm_tdm_wmma<<<gemmGridN, blk, 0, stream>>>(h1bf, w2t, xw, N, HD);
    agg_init_kernel<<<nc, blk, 0, stream>>>(xw, dinv, b2, agg, N);
    scatter_kernel<<<ec, blk, 0, stream>>>(esrc, edst, dinv, xw, agg, E);
    relu_kernel<<<nc, blk, 0, stream>>>(agg, N);                // h2 (f32)

    // ---- head ----
    gather_roots_bf16_kernel<<<dim3((G * (IND / 4) + TB - 1) / TB), blk, 0, stream>>>(x, roots, xrbf, G);
    gemm_tdm_wmma<<<gemmGridG, blk, 0, stream>>>(xrbf, wrt, xrw, G, IND);
    pool_kernel<<<dim3(G), dim3(HD), 0, stream>>>(agg, roots, hpool, N, G);
    final_kernel<<<dim3((G * 2 + TB - 1) / TB), blk, 0, stream>>>(hpool, xrw, br, Wo, bo, out, G);
}
